// KLAttention_33964601377282
// MI455X (gfx1250) — matrix operand order repaired and checked by reading, not compiled, not run on hardware
//
#include <hip/hip_runtime.h>

#define BB 32
#define SS 2048
#define DD 512
#define HH 512

typedef __attribute__((ext_vector_type(16))) _Float16     v16h;
typedef __attribute__((ext_vector_type(8)))  float        v8f;
typedef __attribute__((ext_vector_type(4)))  unsigned int u32x4;

// Global-address-space pointer types: force global_load_b128 (LOADcnt only)
// instead of flat_load_b128 (LOADcnt + DScnt) when inference fails.
typedef const _Float16 __attribute__((address_space(1))) gcf16;
typedef const u32x4    __attribute__((address_space(1))) gcu4;

__device__ __forceinline__ const gcf16* as_global(const _Float16* p) {
    return (const gcf16*)(unsigned long long)p;
}

union AFrag { v16h v; unsigned u[8]; };
union BFrag { v16h v; u32x4 q[2]; };

// A-matrix 16x32 f16 fragment gather from LDS (contiguous k = 16h + s, same order as load_b_glb):
// lane l: M = l%16, h = l/16; dword v<4 : K = h*16 + 2v, 2v+1
//                             dword v>=4: K = h*16 + 8 + 2(v-4), +1
__device__ __forceinline__ v16h load_a_lds(const _Float16* __restrict__ buf,
                                           int ld, int m, int kbase, int h) {
    AFrag a;
    const _Float16* row = buf + (size_t)m * ld + kbase + h * 16;
#pragma unroll
    for (int v = 0; v < 4; ++v) a.u[v]     = *(const unsigned*)(row + 2 * v);
#pragma unroll
    for (int v = 0; v < 4; ++v) a.u[4 + v] = *(const unsigned*)(row + 8 + 2 * v);
    return a.v;
}

// B-matrix 32x16 f16 fragment: lane l: N = l%16, h = l/16; element e: K = h*16+e
// -> 32 contiguous bytes per lane when the "N" rows are K-major in memory.
__device__ __forceinline__ v16h load_b_glb(const _Float16* base, int ld, int n,
                                           int kbase, int h) {
    BFrag b;
    const gcf16* p = as_global(base) + (size_t)n * ld + kbase + h * 16;
    b.q[0] = *(const gcu4*)(p);
    b.q[1] = *(const gcu4*)(p + 8);
    return b.v;
}

__device__ __forceinline__ v8f wmma_f16(v16h a, v16h b, v8f c) {
    return __builtin_amdgcn_wmma_f32_16x16x32_f16(false, a, false, b,
                                                  (short)0, c, false, false);
}

// ---------------------------------------------------------------------------
// Kernel 0: W [D][H] f32 -> WT [H][D] f16 (so B-fragments are contiguous)
// ---------------------------------------------------------------------------
__global__ __launch_bounds__(256) void wt_kernel(const float* __restrict__ W,
                                                 _Float16* __restrict__ WT) {
    int idx = blockIdx.x * 256 + threadIdx.x;
    if (idx < DD * HH) {
        int h = idx % HH;
        int d = idx / HH;
        WT[(size_t)h * DD + d] = (_Float16)W[(size_t)d * HH + h];
    }
}

// ---------------------------------------------------------------------------
// Kernel 1: projections. Block = 16 rows of x:
//   p   = softmax(x@Wq)      -> f16  [B*S][H]
//   lq  = log_softmax(x@Wk)  -> f16  [B*S][H]
//   vT  = (x@Wv)^T per batch -> f16  [B][H][S]
// 8 waves, wave w owns output columns [w*64, w*64+64) = 4 WMMA N-tiles.
// ---------------------------------------------------------------------------
__global__ __launch_bounds__(256) void proj_kernel(
    const float* __restrict__ x,
    const _Float16* __restrict__ WTq, const _Float16* __restrict__ WTk,
    const _Float16* __restrict__ WTv,
    _Float16* __restrict__ Pp, _Float16* __restrict__ Lq,
    _Float16* __restrict__ VT) {
    __shared__ _Float16 xt[16 * 512];     // 16 KB: x tile in f16
    __shared__ float acc_s[16 * 512];     // 32 KB: f32 accum staging

    const int tid  = threadIdx.x;
    const int wave = tid >> 5, lane = tid & 31;
    const int h = lane >> 4, nn = lane & 15;
    const int blk = blockIdx.x;
    const int b   = blk / (SS / 16);
    const int s0  = (blk % (SS / 16)) * 16;
    const size_t rowbase = (size_t)b * SS + s0;

    // cooperative load x tile (f32 -> f16)
    {
        int r  = tid >> 4;
        int c0 = (tid & 15) * 32;
        const float* src = x + (rowbase + r) * DD + c0;
#pragma unroll
        for (int i = 0; i < 32; ++i) xt[r * 512 + c0 + i] = (_Float16)src[i];
    }
    __syncthreads();

    const _Float16* wts[3] = {WTq, WTk, WTv};
    const int hbase = wave * 64;

    for (int m = 0; m < 3; ++m) {
        const _Float16* WT = wts[m];
        v8f acc[4];
#pragma unroll
        for (int t = 0; t < 4; ++t)
#pragma unroll
            for (int r = 0; r < 8; ++r) acc[t][r] = 0.0f;

#pragma unroll 4
        for (int ks = 0; ks < 16; ++ks) {
            const int kb = ks * 32;
            v16h a = load_a_lds(xt, 512, nn, kb, h);
#pragma unroll
            for (int t = 0; t < 4; ++t) {
                v16h bf = load_b_glb(WT, DD, hbase + t * 16 + nn, kb, h);
                acc[t] = wmma_f16(a, bf, acc[t]);
            }
        }

        // spill accumulators: C layout lane l: N = l%16, VGPR r: M = r + h*8
#pragma unroll
        for (int t = 0; t < 4; ++t)
#pragma unroll
            for (int r = 0; r < 8; ++r)
                acc_s[(r + h * 8) * 512 + hbase + t * 16 + nn] = acc[t][r];
        __syncthreads();

        if (m < 2) {
            // wave w owns rows {2w, 2w+1}; lanes 0-15 row 2w, 16-31 row 2w+1
            const int row = 2 * wave + h;
            const float* arow = acc_s + row * 512;
            float mx = -__builtin_inff();
#pragma unroll
            for (int i = 0; i < 32; ++i) mx = fmaxf(mx, arow[nn + 16 * i]);
#pragma unroll
            for (int msk = 8; msk >= 1; msk >>= 1)
                mx = fmaxf(mx, __shfl_xor(mx, msk, 16));
            float sum = 0.0f;
#pragma unroll
            for (int i = 0; i < 32; ++i) sum += __expf(arow[nn + 16 * i] - mx);
#pragma unroll
            for (int msk = 8; msk >= 1; msk >>= 1)
                sum += __shfl_xor(sum, msk, 16);
            if (m == 0) {
                const float inv = 1.0f / sum;
                _Float16* dst = Pp + (rowbase + row) * HH;
#pragma unroll
                for (int i = 0; i < 32; ++i) {
                    int c = nn + 16 * i;
                    dst[c] = (_Float16)(__expf(arow[c] - mx) * inv);
                }
            } else {
                const float lse = mx + __logf(sum);
                _Float16* dst = Lq + (rowbase + row) * HH;
#pragma unroll
                for (int i = 0; i < 32; ++i) {
                    int c = nn + 16 * i;
                    dst[c] = (_Float16)(arow[c] - lse);
                }
            }
        } else {
            // v: store transposed VT[b][h][s] for contiguous PV B-fragments
            int r  = tid >> 4;
            int c0 = (tid & 15) * 32;
#pragma unroll
            for (int i = 0; i < 32; ++i) {
                int c = c0 + i;
                VT[((size_t)b * HH + c) * SS + (s0 + r)] =
                    (_Float16)acc_s[r * 512 + c];
            }
        }
        __syncthreads();
    }
}

// ---------------------------------------------------------------------------
// Kernel 2: flash attention with score_ij = p_i . logq_j (neg-entropy term
// cancels under softmax shift-invariance). Block = 16 query rows.
// j-chunks of 128: score phase -> wave w computes one 16x16 tile for
// j in [jc+16w, jc+16w+16); PV phase -> wave w owns 64 output h-columns.
// ---------------------------------------------------------------------------
__global__ __launch_bounds__(256) void attn_kernel(
    const _Float16* __restrict__ Pp, const _Float16* __restrict__ Lq,
    const _Float16* __restrict__ VT, float* __restrict__ out) {
    __shared__ _Float16 ptile[16 * 512];   // 16 KB: p rows (A operand, reused)
    __shared__ float    scores[16 * 128];  //  8 KB
    __shared__ _Float16 pprob[16 * 128];   //  4 KB: exp(scores) as next A
    __shared__ float row_m[16], row_l[16], row_alpha[16];

    const int tid  = threadIdx.x;
    const int wave = tid >> 5, lane = tid & 31;
    const int h = lane >> 4, nn = lane & 15;
    const int blk = blockIdx.x;
    const int b   = blk / (SS / 16);
    const int s0  = (blk % (SS / 16)) * 16;
    const size_t rowbase = (size_t)b * SS + s0;

    // load p tile once (vectorized, 64 B/thread)
    {
        int r  = tid >> 4;
        int c0 = (tid & 15) * 32;
        const gcu4* s4 = (const gcu4*)(as_global(Pp) + (rowbase + r) * HH + c0);
        u32x4* d4 = (u32x4*)(ptile + r * 512 + c0);
#pragma unroll
        for (int i = 0; i < 4; ++i) d4[i] = s4[i];
    }
    if (tid < 16) { row_m[tid] = -__builtin_inff(); row_l[tid] = 0.0f; }
    __syncthreads();

    v8f oacc[4];
#pragma unroll
    for (int t = 0; t < 4; ++t)
#pragma unroll
        for (int r = 0; r < 8; ++r) oacc[t][r] = 0.0f;

    const int hbase = wave * 64;
    const _Float16* lqb = Lq + (size_t)b * SS * HH;
    const _Float16* vtb = VT + (size_t)b * HH * SS;

    for (int jc = 0; jc < SS; jc += 128) {
        // ---- 1. score tile: cross = p (16xH) . logq^T (Hx16) ----
        v8f sc;
#pragma unroll
        for (int r = 0; r < 8; ++r) sc[r] = 0.0f;
        const _Float16* brow = lqb + (size_t)(jc + wave * 16) * HH;
        if (jc + 128 < SS)  // warm L2 for next chunk's K rows
            __builtin_prefetch(lqb + (size_t)(jc + 128 + wave * 16 + nn) * HH, 0, 1);
#pragma unroll 4
        for (int ks = 0; ks < 16; ++ks) {
            v16h a  = load_a_lds(ptile, 512, nn, ks * 32, h);
            v16h bf = load_b_glb(brow, HH, nn, ks * 32, h);
            sc = wmma_f16(a, bf, sc);
        }
#pragma unroll
        for (int r = 0; r < 8; ++r)
            scores[(r + h * 8) * 128 + wave * 16 + nn] = sc[r];
        __syncthreads();

        // ---- 2. online softmax stats: wave w handles rows {2w, 2w+1} ----
        {
            const int row = 2 * wave + h;
            float* srow = scores + row * 128;
            const float mprev = row_m[row];
            float mx = mprev;
#pragma unroll
            for (int i = 0; i < 8; ++i) mx = fmaxf(mx, srow[nn + 16 * i]);
#pragma unroll
            for (int msk = 8; msk >= 1; msk >>= 1)
                mx = fmaxf(mx, __shfl_xor(mx, msk, 16));
            const float alpha = __expf(mprev - mx);
            float sum = 0.0f;
            _Float16* prow = pprob + row * 128;
#pragma unroll
            for (int i = 0; i < 8; ++i) {
                float e = __expf(srow[nn + 16 * i] - mx);
                prow[nn + 16 * i] = (_Float16)e;
                sum += e;
            }
#pragma unroll
            for (int msk = 8; msk >= 1; msk >>= 1)
                sum += __shfl_xor(sum, msk, 16);
            if (nn == 0) {
                row_m[row]     = mx;
                row_alpha[row] = alpha;
                row_l[row]     = row_l[row] * alpha + sum;
            }
        }
        __syncthreads();

        // ---- 3. rescale + PV: oacc = oacc*alpha + prob (16x128).v (128x64) ----
        float al[8];
#pragma unroll
        for (int r = 0; r < 8; ++r) al[r] = row_alpha[r + h * 8];
#pragma unroll
        for (int t = 0; t < 4; ++t)
#pragma unroll
            for (int r = 0; r < 8; ++r) oacc[t][r] *= al[r];

#pragma unroll 2
        for (int ks = 0; ks < 4; ++ks) {
            v16h a = load_a_lds(pprob, 128, nn, ks * 32, h);
#pragma unroll
            for (int t = 0; t < 4; ++t) {
                v16h bf = load_b_glb(vtb, SS, hbase + t * 16 + nn,
                                     jc + ks * 32, h);
                oacc[t] = wmma_f16(a, bf, oacc[t]);
            }
        }
        __syncthreads();  // protect scores/pprob reuse next chunk
    }

    // ---- finalize: divide by softmax denominator, store f32 ----
    float ls[8];
#pragma unroll
    for (int r = 0; r < 8; ++r) ls[r] = 1.0f / row_l[r + h * 8];
#pragma unroll
    for (int t = 0; t < 4; ++t)
#pragma unroll
        for (int r = 0; r < 8; ++r)
            out[(rowbase + r + h * 8) * HH + hbase + t * 16 + nn] =
                oacc[t][r] * ls[r];
}

// ---------------------------------------------------------------------------
extern "C" void kernel_launch(void* const* d_in, const int* in_sizes, int n_in,
                              void* d_out, int out_size, void* d_ws,
                              size_t ws_size, hipStream_t stream) {
    (void)in_sizes; (void)n_in; (void)out_size; (void)ws_size;
    const float* x  = (const float*)d_in[0];
    const float* Wq = (const float*)d_in[1];
    const float* Wk = (const float*)d_in[2];
    const float* Wv = (const float*)d_in[3];
    float* out = (float*)d_out;

    char* ws = (char*)d_ws;
    const size_t MB64 = (size_t)64 << 20;  // B*S*H * 2B = exactly 64 MiB
    _Float16* Pp  = (_Float16*)(ws);              // p  = softmax(q)   f16
    _Float16* Lq  = (_Float16*)(ws + MB64);       // log_softmax(k)    f16
    _Float16* VT  = (_Float16*)(ws + 2 * MB64);   // v transposed      f16
    _Float16* WTq = (_Float16*)(ws + 3 * MB64);   // weight transposes f16
    _Float16* WTk = WTq + (size_t)DD * HH;
    _Float16* WTv = WTk + (size_t)DD * HH;

    const int nW = DD * HH;
    wt_kernel<<<(nW + 255) / 256, 256, 0, stream>>>(Wq, WTq);
    wt_kernel<<<(nW + 255) / 256, 256, 0, stream>>>(Wk, WTk);
    wt_kernel<<<(nW + 255) / 256, 256, 0, stream>>>(Wv, WTv);

    const int nblk = BB * SS / 16;  // 4096
    proj_kernel<<<nblk, 256, 0, stream>>>(x, WTq, WTk, WTv, Pp, Lq, VT);
    attn_kernel<<<nblk, 256, 0, stream>>>(Pp, Lq, VT, out);
}
